// SelfAttentionHead_19816979104169
// MI455X (gfx1250) — compile-verified, matrix operand order checked by reading, not run on hardware
//
#include <hip/hip_runtime.h>
#include <hip/hip_bf16.h>
#include <math.h>

#define B_   4
#define S_   4096
#define D_   512
#define BS_  (B_ * S_)
#define BSD_ ((size_t)B_ * S_ * D_)
#define NEGV (-1e9f)
#define SCALE_ 0.04419417382415922f  // 1/sqrt(512)

typedef __attribute__((ext_vector_type(16))) __bf16 v16bf;
typedef __attribute__((ext_vector_type(8)))  float v8f;

struct FragU { uint4 lo, hi; };
union FragCast { FragU u; v16bf v; };

__device__ __forceinline__ unsigned short f2bf(float f) {
  unsigned int u = __float_as_uint(f);
  u += 0x7FFFu + ((u >> 16) & 1u);   // round-to-nearest-even
  return (unsigned short)(u >> 16);
}

// Load one 16x32 bf16 A/B fragment. For a row-major [rows x ld] matrix with
// tile origin (row0, k0): lane = grp*16+lo -> row row0+lo,
// halves: grp==0 -> k0+{0..7, 16..23}, grp==1 -> k0+{8..15, 24..31}
// (CDNA5 ISA 16-bit A/B VGPR layout; B operands come from the transposed
// matrix stored row-major, so the same pattern serves both.)
__device__ __forceinline__ const unsigned short*
frag_ptr(const unsigned short* base, long ld, int row0, int k0) {
  int lane = threadIdx.x & 31;
  int lo = lane & 15, grp = lane >> 4;
  return base + (long)(row0 + lo) * ld + (k0 + grp * 8);
}

__device__ __forceinline__ v16bf load_frag(const unsigned short* p) {
  FragCast fc;
  fc.u.lo = *(const uint4*)(p);
  fc.u.hi = *(const uint4*)(p + 16);
  return fc.v;
}

// ---------------------------------------------------------------------------
// Kernel 1: QKV projection.  out = x @ W^T + b  (W stored [out,in] row-major)
// grid = (BS/128, D/64, 3), block = 256 (8 waves). Wave w: rows w*16..+15,
// all 64 cols of the block tile.
// ---------------------------------------------------------------------------
#define TM 128
#define TN 64
#define TK 32
#define XLD 40   // padded LDS stride (halves)
#define WLD 40

__global__ __launch_bounds__(256) void qkv_proj_kernel(
    const float* __restrict__ x,
    const float* __restrict__ Wq, const float* __restrict__ Wk, const float* __restrict__ Wv,
    const float* __restrict__ bq, const float* __restrict__ bk, const float* __restrict__ bv,
    float* __restrict__ keys_f32, float* __restrict__ vals_f32,
    unsigned short* __restrict__ qbf, unsigned short* __restrict__ kbf,
    unsigned short* __restrict__ vtbf)
{
  __shared__ unsigned short xs[TM * XLD];
  __shared__ unsigned short ws[TN * WLD];

  const int z = blockIdx.z;
  const float* W    = (z == 0) ? Wq : (z == 1) ? Wk : Wv;
  const float* bias = (z == 0) ? bq : (z == 1) ? bk : bv;

  const int m0 = blockIdx.x * TM;
  const int n0 = blockIdx.y * TN;
  const int tid  = threadIdx.x;
  const int wv   = tid >> 5;
  const int lane = tid & 31;
  const int lo = lane & 15, grp = lane >> 4;

  v8f acc[4] = {};

  for (int k0 = 0; k0 < D_; k0 += TK) {
    // stage X tile 128x32 (f32 -> bf16)
#pragma unroll
    for (int i = 0; i < 4; ++i) {
      int c = tid + i * 256;                  // 1024 chunks of 4 floats
      int r = c >> 3, cc = (c & 7) * 4;
      float4 v = *(const float4*)(x + (long)(m0 + r) * D_ + k0 + cc);
      uint2 p;
      p.x = (unsigned)f2bf(v.x) | ((unsigned)f2bf(v.y) << 16);
      p.y = (unsigned)f2bf(v.z) | ((unsigned)f2bf(v.w) << 16);
      *(uint2*)&xs[r * XLD + cc] = p;
    }
    // stage W tile 64x32
#pragma unroll
    for (int i = 0; i < 2; ++i) {
      int c = tid + i * 256;                  // 512 chunks
      int r = c >> 3, cc = (c & 7) * 4;
      float4 v = *(const float4*)(W + (long)(n0 + r) * D_ + k0 + cc);
      uint2 p;
      p.x = (unsigned)f2bf(v.x) | ((unsigned)f2bf(v.y) << 16);
      p.y = (unsigned)f2bf(v.z) | ((unsigned)f2bf(v.w) << 16);
      *(uint2*)&ws[r * WLD + cc] = p;
    }
    __syncthreads();

    v16bf a = load_frag(frag_ptr(xs, XLD, wv * 16, 0));
#pragma unroll
    for (int t = 0; t < 4; ++t) {
      v16bf bm = load_frag(frag_ptr(ws, WLD, t * 16, 0));
      acc[t] = __builtin_amdgcn_wmma_f32_16x16x32_bf16(
          false, a, false, bm, (short)0, acc[t], false, false);
    }
    __syncthreads();
  }

  // epilogue: bias + store (Q: bf16 only; K: f32 + bf16; V: f32 + bf16^T)
#pragma unroll
  for (int t = 0; t < 4; ++t) {
    int col = n0 + t * 16 + lo;
    float badd = bias[col];
#pragma unroll
    for (int r = 0; r < 8; ++r) {
      int  M = r + 8 * grp;
      long row_g = (long)m0 + wv * 16 + M;
      float val = acc[t][r] + badd;
      long flat = row_g * D_ + col;
      if (z == 0) {
        qbf[flat] = f2bf(val);
      } else if (z == 1) {
        keys_f32[flat] = val;
        kbf[flat] = f2bf(val);
      } else {
        vals_f32[flat] = val;
        long bb = row_g / S_, ss = row_g % S_;
        vtbf[bb * (long)D_ * S_ + (long)col * S_ + ss] = f2bf(val);  // V^T
      }
    }
  }
}

// ---------------------------------------------------------------------------
// Kernel 2: flash attention.  grid = (S/32, B), block = 256 (8 waves).
// Per WG: 32 queries (2 subtiles of 16). Wave w = (qw = w>>2, dw = w&3):
// Phase A: partial S for its query subtile over head-dims dw*128..+127,
// Phase B: online softmax across the 32x32 score block (all 256 threads),
// Phase C: O += P @ V with each wave owning 128 head-dims (via V^T rows).
// 32-query tiles halve L2 K/V re-read traffic vs 16-query tiles while
// keeping 64 accumulator VGPRs per lane and 512 WGs for load balance.
// ---------------------------------------------------------------------------
#define QLD 520   // 512 + 8 pad (halves)
#define PLD 40

__global__ __launch_bounds__(256) void attn_kernel(
    const unsigned short* __restrict__ qbf,
    const unsigned short* __restrict__ kbf,
    const unsigned short* __restrict__ vtbf,
    const float* __restrict__ pad_mask,
    float* __restrict__ att_out)
{
  __shared__ unsigned short qs[32 * QLD];
  __shared__ float sred[2][4][16][33];
  __shared__ unsigned short pls[2][16 * PLD];
  __shared__ float mArr[32], lArr[32], scArr[32];

  const int q0 = blockIdx.x * 32;
  const int b  = blockIdx.y;
  const int tid  = threadIdx.x;
  const int wv   = tid >> 5;
  const int qw   = wv >> 2;      // query subtile 0..1
  const int dw   = wv & 3;       // head-dim slice 0..3
  const int lane = tid & 31;
  const int lo = lane & 15, grp = lane >> 4;

  const unsigned short* qg  = qbf  + (long)b * S_ * D_;
  const unsigned short* kg  = kbf  + (long)b * S_ * D_;
  const unsigned short* vtg = vtbf + (long)b * D_ * S_;
  const float* pm = pad_mask + (long)b * S_;

  // stage Q tile (32 x 512 bf16) once
#pragma unroll
  for (int i = 0; i < 8; ++i) {
    int c = tid + i * 256;                    // 2048 chunks of 8 halves
    int r = c >> 6, c8 = (c & 63) * 8;
    *(uint4*)&qs[r * QLD + c8] = *(const uint4*)(qg + (long)(q0 + r) * D_ + c8);
  }
  if (tid < 32) { mArr[tid] = -INFINITY; lArr[tid] = 0.0f; scArr[tid] = 0.0f; }
  __syncthreads();

  v8f o[8] = {};
  const int nkb = (q0 + 32 + 31) / 32;

  for (int kb = 0; kb < nkb; ++kb) {
    const int kb0 = kb * 32;

    // prefetch next key block (emits global_prefetch_b8)
    if (kb + 1 < nkb)
      __builtin_prefetch(kg + (long)(kb0 + 32) * D_ + (long)tid * 64, 0, 1);

    // Phase A: partial S (queries qw*16..+15) over dims [dw*128, dw*128+128)
    v8f s0 = {}, s1 = {};
#pragma unroll
    for (int i = 0; i < 4; ++i) {
      int kq = dw * 128 + i * 32;
      v16bf a  = load_frag(frag_ptr(qs, QLD, qw * 16, kq));
      v16bf b0 = load_frag(frag_ptr(kg, D_, kb0,      kq));
      v16bf b1 = load_frag(frag_ptr(kg, D_, kb0 + 16, kq));
      s0 = __builtin_amdgcn_wmma_f32_16x16x32_bf16(false, a, false, b0, (short)0, s0, false, false);
      s1 = __builtin_amdgcn_wmma_f32_16x16x32_bf16(false, a, false, b1, (short)0, s1, false, false);
    }
#pragma unroll
    for (int r = 0; r < 8; ++r) {
      sred[qw][dw][r + 8 * grp][lo]      = s0[r];
      sred[qw][dw][r + 8 * grp][16 + lo] = s1[r];
    }
    __syncthreads();

    // Phase B: reduce partials, mask, online-softmax update (32 rows x 32 cols)
    {
      int row  = tid >> 3;           // 0..31 global query row in tile
      int g    = row >> 4;           // query subtile
      int rloc = row & 15;
      int c0   = (tid & 7) * 4;
      float v[4];
      float mloc = -INFINITY;
#pragma unroll
      for (int j = 0; j < 4; ++j) {
        int c = c0 + j;
        float s = sred[g][0][rloc][c] + sred[g][1][rloc][c]
                + sred[g][2][rloc][c] + sred[g][3][rloc][c];
        s *= SCALE_;
        int kglob = kb0 + c;
        s += pm[kglob];                       // additive padding mask
        if (kglob > q0 + row) s += NEGV;      // additive causal mask
        v[j] = s;
        mloc = fmaxf(mloc, s);
      }
      mloc = fmaxf(mloc, __shfl_xor(mloc, 1, 32));
      mloc = fmaxf(mloc, __shfl_xor(mloc, 2, 32));
      mloc = fmaxf(mloc, __shfl_xor(mloc, 4, 32));
      float m_old = mArr[row];
      float m_new = fmaxf(m_old, mloc);
      float psum = 0.0f;
#pragma unroll
      for (int j = 0; j < 4; ++j) {
        float p = __expf(v[j] - m_new);
        psum += p;
        pls[g][rloc * PLD + c0 + j] = f2bf(p);
      }
      psum += __shfl_xor(psum, 1, 32);
      psum += __shfl_xor(psum, 2, 32);
      psum += __shfl_xor(psum, 4, 32);
      float sc = __expf(m_old - m_new);
      if ((tid & 7) == 0) {
        mArr[row]  = m_new;
        lArr[row]  = lArr[row] * sc + psum;
        scArr[row] = sc;
      }
    }
    __syncthreads();

    // Phase C: O = O*sc + P @ V  (wave owns dims dw*128..+127 via V^T rows)
    {
      float sc[8];
#pragma unroll
      for (int r = 0; r < 8; ++r) sc[r] = scArr[qw * 16 + r + 8 * grp];
      v16bf a = load_frag(frag_ptr(pls[qw], PLD, 0, 0));
#pragma unroll
      for (int t = 0; t < 8; ++t) {
#pragma unroll
        for (int r = 0; r < 8; ++r) o[t][r] *= sc[r];
        v16bf bm = load_frag(frag_ptr(vtg, S_, dw * 128 + t * 16, kb0));
        o[t] = __builtin_amdgcn_wmma_f32_16x16x32_bf16(
            false, a, false, bm, (short)0, o[t], false, false);
      }
    }
    __syncthreads();
  }

  // epilogue: normalize and store
  float linv[8];
#pragma unroll
  for (int r = 0; r < 8; ++r) linv[r] = 1.0f / lArr[qw * 16 + r + 8 * grp];
  float* outb = att_out + (long)b * S_ * D_;
#pragma unroll
  for (int t = 0; t < 8; ++t) {
    int col = dw * 128 + t * 16 + lo;
#pragma unroll
    for (int r = 0; r < 8; ++r) {
      int row = q0 + qw * 16 + r + 8 * grp;
      outb[(long)row * D_ + col] = o[t][r] * linv[r];
    }
  }
}

// ---------------------------------------------------------------------------
extern "C" void kernel_launch(void* const* d_in, const int* in_sizes, int n_in,
                              void* d_out, int out_size, void* d_ws, size_t ws_size,
                              hipStream_t stream) {
  const float* x   = (const float*)d_in[0];
  const float* pad = (const float*)d_in[1];
  // d_in[2] = subsq_mask: reconstructed analytically (triu -1e9)
  const float* Wq = (const float*)d_in[3];
  const float* bq = (const float*)d_in[4];
  const float* Wk = (const float*)d_in[5];
  const float* bk = (const float*)d_in[6];
  const float* Wv = (const float*)d_in[7];
  const float* bv = (const float*)d_in[8];

  float* att_out = (float*)d_out;              // outputs: (att_out, keys, values)
  float* keys    = att_out + BSD_;
  float* vals    = att_out + 2 * BSD_;

  unsigned short* qbf  = (unsigned short*)d_ws;       // bf16 Q  [B,S,D]
  unsigned short* kbf  = qbf + BSD_;                  // bf16 K  [B,S,D]
  unsigned short* vtbf = kbf + BSD_;                  // bf16 V^T [B,D,S]

  dim3 g1(BS_ / TM, D_ / TN, 3), blk1(256);
  qkv_proj_kernel<<<g1, blk1, 0, stream>>>(x, Wq, Wk, Wv, bq, bk, bv,
                                           keys, vals, qbf, kbf, vtbf);

  dim3 g2(S_ / 32, B_), blk2(256);
  attn_kernel<<<g2, blk2, 0, stream>>>(qbf, kbf, vtbf, pad, att_out);
}
